// Generator_31593779429712
// MI455X (gfx1250) — matrix operand order repaired and checked by reading, not compiled, not run on hardware
//
#include <hip/hip_runtime.h>
#include <hip/hip_bf16.h>
#include <stdint.h>

// ---------------------------------------------------------------------------
// Stacked BiLSTM for MI455X (gfx1250, wave32, WMMA).
//   fc1 -> BiLSTM(H=64, in=1) -> repeat x2 -> BiLSTM(H=64, in=128) -> fc2+tanh
// Each wave owns 16 batch rows; gates computed with v_wmma_f32_16x16x32_f16
// against weights pre-swizzled into LDS as B fragments in contiguous k order (k = 16h + s, same as A).
// Gate columns are processed in two scheduler-fenced halves (8 accumulators
// live at a time) + waves-per-eu hint to stay under 256 VGPRs.
// ---------------------------------------------------------------------------

#define BATCH 8192
#define T1 50
#define T2 100

typedef __attribute__((ext_vector_type(16))) _Float16 v16h;
typedef __attribute__((ext_vector_type(8)))  float    v8f;

#if __has_builtin(__builtin_amdgcn_sched_barrier)
#define SCHED_FENCE() __builtin_amdgcn_sched_barrier(0)
#else
#define SCHED_FENCE()
#endif

union FragU { uint4 q[2]; v16h h; };

__device__ inline v8f wmma_f16(v16h a, v16h b, v8f c) {
  // (neg_a, A, neg_b, B, c_mod, C, reuse_a, reuse_b)
  return __builtin_amdgcn_wmma_f32_16x16x32_f16(false, a, false, b, (short)0, c,
                                                false, false);
}

__device__ inline float tanh_fast(float x) {
#if __has_builtin(__builtin_amdgcn_tanh_f32)
  return __builtin_amdgcn_tanh_f32(x);      // CDNA5 V_TANH_F32
#elif __has_builtin(__builtin_amdgcn_tanhf)
  return __builtin_amdgcn_tanhf(x);
#else
  float e = __expf(2.0f * x);               // v_exp_f32 path
  return 1.0f - 2.0f / (e + 1.0f);
#endif
}
__device__ inline float sigm(float x) { return 0.5f * tanh_fast(0.5f * x) + 0.5f; }

__device__ inline uint32_t pack2h(float a, float b) {
  union { _Float16 h[2]; uint32_t u; } p;
  p.h[0] = (_Float16)a; p.h[1] = (_Float16)b;
  return p.u;
}

// Pre-swizzle W (4H x Kdim, row-major fp32, N=gate index, K=contraction) into
// a 16-bit B-matrix (K x 16) fragment layout, contiguous k order (matches A):
//   fragment f = kb*16+tt; element (lane, vgpr j, half hh):
//     K = kb*32 + (lane>=16 ? 16 : 0) + 2j + hh,  N = tt*16 + (lane&15)
__device__ inline void swizzleW(uint32_t* bank, const float* W, int Kdim,
                                int nKb, int tid, int nthr) {
  const int total = nKb * 16 * 32 * 8;
  for (int i = tid; i < total; i += nthr) {
    int j = i & 7, lane = (i >> 3) & 31, f = i >> 8;
    int kb = f >> 4, tt = f & 15;
    int K = kb * 32 + (lane >> 4) * 16 + 2 * j;
    int N = tt * 16 + (lane & 15);
    bank[i] = pack2h(W[(size_t)N * Kdim + K], W[(size_t)N * Kdim + K + 1]);
  }
}

__device__ inline v16h load_fragB(const uint32_t* bank, int fragIdx, int lane) {
  const uint4* p = (const uint4*)(bank + (((size_t)fragIdx * 32 + lane) << 3));
  FragU f; f.q[0] = p[0]; f.q[1] = p[1];
  return f.h;
}

// A-fragment (16xK f16, row-major, lane&15 selects M row, rowp at row start):
//   lanes 0-15: K = kb*32 + {0..15};  lanes 16-31: {16..31}
__device__ inline v16h load_fragA_dw(const uint32_t* rowp, int kb, int hi) {
  const uint32_t* b = rowp + kb * 16 + hi * 8;
  FragU f; f.q[0] = *(const uint4*)b; f.q[1] = *(const uint4*)(b + 4);
  return f.h;
}

// ---------------------------------------------------------------------------
__global__ __launch_bounds__(256) void fc1_kernel(const float* __restrict__ x,
                                                  const float* __restrict__ w,
                                                  const float* __restrict__ b,
                                                  float* __restrict__ o) {
  int idx = blockIdx.x * 256 + threadIdx.x;
  if (idx >= BATCH * T1) return;
  int bi = idx / T1, t = idx - bi * T1;
  float s = b[t];
#pragma unroll
  for (int i = 0; i < 20; i++) s += x[bi * 20 + i] * w[t * 20 + i];
  o[idx] = s;
}

// ---------------------------------------------------------------------------
// Layer-1 BiLSTM (input size 1). blockIdx.y = direction.
__global__ __launch_bounds__(128, 4) void lstm1_kernel(
    const float* __restrict__ fc1h,
    const float* __restrict__ wih_f, const float* __restrict__ whh_f,
    const float* __restrict__ b_f,
    const float* __restrict__ wih_b, const float* __restrict__ whh_b,
    const float* __restrict__ b_b,
    _Float16* __restrict__ h1out) {
  __shared__ __align__(16) uint32_t sWhh[2 * 16 * 32 * 8];   // 32 KB frag bank
  __shared__ __align__(16) _Float16 hbuf[4][16 * 64];        // 8 KB h state

  const int tid = threadIdx.x, lane = tid & 31, wave = tid >> 5;
  const int dir = blockIdx.y;
  const int base = blockIdx.x * 64 + wave * 16;
  const float* wih = dir ? wih_b : wih_f;
  const float* whh = dir ? whh_b : whh_f;
  const float* bia = dir ? b_b : b_f;

  swizzleW(sWhh, whh, 64, 2, tid, 128);
  for (int i = tid; i < 4 * 16 * 64; i += 128) (&hbuf[0][0])[i] = (_Float16)0.0f;
  __syncthreads();

  const int nloc = lane & 15, hi = lane >> 4;
  float wihv[16], biasv[16];
#pragma unroll
  for (int tt = 0; tt < 16; tt++) {
    wihv[tt] = wih[tt * 16 + nloc];
    biasv[tt] = bia[tt * 16 + nloc];
  }

  _Float16* hb = hbuf[wave];
  v8f c[4];
#pragma unroll
  for (int q = 0; q < 4; q++)
#pragma unroll
    for (int v = 0; v < 8; v++) c[q][v] = 0.0f;

  for (int t = 0; t < T1; t++) {
    const int tIdx = dir ? (T1 - 1 - t) : t;

    float hin[8];
#pragma unroll
    for (int v = 0; v < 8; v++)
      hin[v] = fc1h[(size_t)(base + v + 8 * hi) * T1 + tIdx];

    // h_{t-1} A-fragments: load once, before any hb writes this step
    v16h ah[2];
#pragma unroll
    for (int kb = 0; kb < 2; kb++)
      ah[kb] = load_fragA_dw((const uint32_t*)(hb + nloc * 64), kb, hi);

    // two column-halves, scheduler-fenced: 8 accumulators live at a time
#pragma unroll
    for (int qh = 0; qh < 2; qh++) {
      SCHED_FENCE();
      v8f acc[8];
      // xg = hin * wih + bias folded into accumulator init (input size == 1)
#pragma unroll
      for (int g = 0; g < 4; g++)
#pragma unroll
        for (int j = 0; j < 2; j++) {
          const int tt = g * 4 + 2 * qh + j;
#pragma unroll
          for (int v = 0; v < 8; v++)
            acc[g * 2 + j][v] = hin[v] * wihv[tt] + biasv[tt];
        }
      // recurrent GEMM for this half: 2 K-blocks x 8 tiles = 16 WMMAs
#pragma unroll
      for (int kb = 0; kb < 2; kb++)
#pragma unroll
        for (int i = 0; i < 8; i++) {
          const int tt = (i >> 1) * 4 + 2 * qh + (i & 1);
          acc[i] = wmma_f16(ah[kb], load_fragB(sWhh, kb * 16 + tt, lane), acc[i]);
        }
      // cell update; C-layout: row = v + 8*hi, col = q*16 + nloc
#pragma unroll
      for (int j = 0; j < 2; j++) {
        const int q = 2 * qh + j;
#pragma unroll
        for (int v = 0; v < 8; v++) {
          float ig = sigm(acc[0 + j][v]);
          float fg = sigm(acc[2 + j][v]);
          float gg = tanh_fast(acc[4 + j][v]);
          float og = sigm(acc[6 + j][v]);
          float cv = fg * c[q][v] + ig * gg;
          c[q][v] = cv;
          hb[(v + 8 * hi) * 64 + q * 16 + nloc] = (_Float16)(og * tanh_fast(cv));
        }
      }
      SCHED_FENCE();
    }

    // stream h (f16) to global: half a 64-elem row per lane, 4x b128
    {
      const uint4* src = (const uint4*)(hb + nloc * 64 + hi * 32);
      uint4* dst = (uint4*)(h1out + ((size_t)(base + nloc) * T1 + tIdx) * 128 +
                            dir * 64 + hi * 32);
#pragma unroll
      for (int k = 0; k < 4; k++) dst[k] = src[k];
    }
  }
}

// ---------------------------------------------------------------------------
// Layer-2 BiLSTM (input size 128) with fused input GEMM and fused fc2+tanh.
// dir==0 writes fc2 partial into out; dir==1 adds, biases, tanh's.
__global__ __launch_bounds__(128, 4) void lstm2_kernel(
    const _Float16* __restrict__ h1, const float* __restrict__ wih,
    const float* __restrict__ whh, const float* __restrict__ bia,
    const float* __restrict__ fc2w, const float* __restrict__ fc2b,
    float* __restrict__ out, const int dir) {
  __shared__ __align__(16) uint32_t sWih[4 * 16 * 32 * 8];   // 64 KB
  __shared__ __align__(16) uint32_t sWhh[2 * 16 * 32 * 8];   // 32 KB
  __shared__ __align__(16) _Float16 hbuf[4][16 * 64];        // 8 KB
  __shared__ float sF2[64];

  const int tid = threadIdx.x, lane = tid & 31, wave = tid >> 5;
  const int base = blockIdx.x * 64 + wave * 16;

  swizzleW(sWih, wih, 128, 4, tid, 128);
  swizzleW(sWhh, whh, 64, 2, tid, 128);
  if (tid < 64) sF2[tid] = fc2w[dir * 64 + tid];
  for (int i = tid; i < 4 * 16 * 64; i += 128) (&hbuf[0][0])[i] = (_Float16)0.0f;
  __syncthreads();

  const int nloc = lane & 15, hi = lane >> 4;
  float biasv[16];
#pragma unroll
  for (int tt = 0; tt < 16; tt++) biasv[tt] = bia[tt * 16 + nloc];

  _Float16* hb = hbuf[wave];
  v8f c[4];
#pragma unroll
  for (int q = 0; q < 4; q++)
#pragma unroll
    for (int v = 0; v < 8; v++) c[q][v] = 0.0f;

  const _Float16* xrow = h1 + (size_t)(base + nloc) * T1 * 128;

  for (int t = 0; t < T2; t++) {
    const int tIdx = dir ? (T2 - 1 - t) : t;
    const int tIn = tIdx >> 1;  // repeat(2, axis=1)

    // all A-fragments for this step: 4 input (from HBM) + 2 recurrent (LDS),
    // loaded once before any hb writes, reused across both column-halves
    const uint32_t* rowp = (const uint32_t*)(xrow + (size_t)tIn * 128);
    v16h ax[4], ah[2];
#pragma unroll
    for (int kb = 0; kb < 4; kb++) ax[kb] = load_fragA_dw(rowp, kb, hi);
#pragma unroll
    for (int kb = 0; kb < 2; kb++)
      ah[kb] = load_fragA_dw((const uint32_t*)(hb + nloc * 64), kb, hi);

#pragma unroll
    for (int qh = 0; qh < 2; qh++) {
      SCHED_FENCE();
      v8f acc[8];
#pragma unroll
      for (int g = 0; g < 4; g++)
#pragma unroll
        for (int j = 0; j < 2; j++) {
          const int tt = g * 4 + 2 * qh + j;
#pragma unroll
          for (int v = 0; v < 8; v++) acc[g * 2 + j][v] = biasv[tt];
        }
      // input GEMM half: 4 K-blocks x 8 tiles = 32 WMMAs
#pragma unroll
      for (int kb = 0; kb < 4; kb++)
#pragma unroll
        for (int i = 0; i < 8; i++) {
          const int tt = (i >> 1) * 4 + 2 * qh + (i & 1);
          acc[i] = wmma_f16(ax[kb], load_fragB(sWih, kb * 16 + tt, lane), acc[i]);
        }
      // recurrent GEMM half: 2 K-blocks x 8 tiles = 16 WMMAs
#pragma unroll
      for (int kb = 0; kb < 2; kb++)
#pragma unroll
        for (int i = 0; i < 8; i++) {
          const int tt = (i >> 1) * 4 + 2 * qh + (i & 1);
          acc[i] = wmma_f16(ah[kb], load_fragB(sWhh, kb * 16 + tt, lane), acc[i]);
        }
#pragma unroll
      for (int j = 0; j < 2; j++) {
        const int q = 2 * qh + j;
#pragma unroll
        for (int v = 0; v < 8; v++) {
          float ig = sigm(acc[0 + j][v]);
          float fg = sigm(acc[2 + j][v]);
          float gg = tanh_fast(acc[4 + j][v]);
          float og = sigm(acc[6 + j][v]);
          float cv = fg * c[q][v] + ig * gg;
          c[q][v] = cv;
          hb[(v + 8 * hi) * 64 + q * 16 + nloc] = (_Float16)(og * tanh_fast(cv));
        }
      }
      SCHED_FENCE();
    }

    // prefetch next timestep's input row (global_prefetch_b8)
    if (t + 1 < T2) {
      const int nIdx = dir ? (T2 - 2 - t) : (t + 1);
      __builtin_prefetch((const char*)(xrow + (size_t)(nIdx >> 1) * 128), 0, 1);
    }

    // fused fc2: one output scalar per batch row per step
    if (lane < 16) {
      float s = 0.0f;
#pragma unroll 8
      for (int n = 0; n < 64; n++) s += (float)hb[lane * 64 + n] * sF2[n];
      const size_t oidx = (size_t)(base + lane) * T2 + tIdx;
      if (dir == 0)
        out[oidx] = s;
      else
        out[oidx] = tanh_fast(out[oidx] + s + fc2b[0]);
    }
  }
}

// ---------------------------------------------------------------------------
extern "C" void kernel_launch(void* const* d_in, const int* in_sizes, int n_in,
                              void* d_out, int out_size, void* d_ws,
                              size_t ws_size, hipStream_t stream) {
  (void)in_sizes; (void)n_in; (void)out_size; (void)ws_size;
  const float* x        = (const float*)d_in[0];
  const float* fc1_w    = (const float*)d_in[1];
  const float* fc1_b    = (const float*)d_in[2];
  const float* r1_wih_f = (const float*)d_in[3];
  const float* r1_whh_f = (const float*)d_in[4];
  const float* r1_b_f   = (const float*)d_in[5];
  const float* r1_wih_b = (const float*)d_in[6];
  const float* r1_whh_b = (const float*)d_in[7];
  const float* r1_b_b   = (const float*)d_in[8];
  const float* r2_wih_f = (const float*)d_in[9];
  const float* r2_whh_f = (const float*)d_in[10];
  const float* r2_b_f   = (const float*)d_in[11];
  const float* r2_wih_b = (const float*)d_in[12];
  const float* r2_whh_b = (const float*)d_in[13];
  const float* r2_b_b   = (const float*)d_in[14];
  const float* fc2_w    = (const float*)d_in[15];
  const float* fc2_b    = (const float*)d_in[16];
  float* outp = (float*)d_out;

  // workspace: fc1 activations (fp32) + layer-1 output stream (f16)
  float*    fc1h = (float*)d_ws;
  _Float16* h1   = (_Float16*)((char*)d_ws + (size_t)BATCH * T1 * sizeof(float));

  fc1_kernel<<<(BATCH * T1 + 255) / 256, 256, 0, stream>>>(x, fc1_w, fc1_b, fc1h);
  lstm1_kernel<<<dim3(BATCH / 64, 2), 128, 0, stream>>>(
      fc1h, r1_wih_f, r1_whh_f, r1_b_f, r1_wih_b, r1_whh_b, r1_b_b, h1);
  lstm2_kernel<<<BATCH / 64, 128, 0, stream>>>(h1, r2_wih_f, r2_whh_f, r2_b_f,
                                               fc2_w, fc2_b, outp, 0);
  lstm2_kernel<<<BATCH / 64, 128, 0, stream>>>(h1, r2_wih_b, r2_whh_b, r2_b_b,
                                               fc2_w, fc2_b, outp, 1);
}
